// RendNet_71657234367218
// MI455X (gfx1250) — matrix operand order repaired and checked by reading, not compiled, not run on hardware
//
#include <hip/hip_runtime.h>

// ---------------------------------------------------------------------------
// Types for CDNA5 WMMA (wave32)
// ---------------------------------------------------------------------------
typedef __attribute__((ext_vector_type(16))) __bf16 v16bf;
typedef __attribute__((ext_vector_type(8)))  float  v8f;

union FragBF { v16bf bf; uint4 q[2]; };

__device__ __forceinline__ unsigned short f2bf(float f) {
  unsigned int u = __float_as_uint(f);
  u += 0x7FFFu + ((u >> 16) & 1u);      // round-to-nearest-even
  return (unsigned short)(u >> 16);
}

__device__ __forceinline__ float bilin_px(const float* __restrict__ ch, int H, int W,
                                          float fx, float fy) {
  int x0 = (int)floorf(fx);
  int y0 = (int)floorf(fy);
  x0 = x0 < 0 ? 0 : (x0 > W - 1 ? W - 1 : x0);
  y0 = y0 < 0 ? 0 : (y0 > H - 1 ? H - 1 : y0);
  int x1 = (x0 + 1 > W - 1) ? W - 1 : x0 + 1;
  int y1 = (y0 + 1 > H - 1) ? H - 1 : y0 + 1;
  float wx = fx - (float)x0;
  float wy = fy - (float)y0;
  float v00 = ch[(size_t)y0 * W + x0];
  float v01 = ch[(size_t)y0 * W + x1];
  float v10 = ch[(size_t)y1 * W + x0];
  float v11 = ch[(size_t)y1 * W + x1];
  float top = v00 + (v01 - v00) * wx;
  float bot = v10 + (v11 - v10) * wx;
  return top + (bot - top) * wy;
}

__device__ __forceinline__ float bilin01(const float* __restrict__ ch, int H, int W,
                                         float px, float py) {
  return bilin_px(ch, H, W, px * (float)(W - 1), py * (float)(H - 1));
}

// ---------------------------------------------------------------------------
// interp2x (align_corners): out [CT, 2H, 2W] from in [CT, H, W]
// ---------------------------------------------------------------------------
__global__ __launch_bounds__(256) void upsample2x(const float* __restrict__ in,
                                                  float* __restrict__ out,
                                                  int CT, int H, int W) {
  int Ho = H * 2, Wo = W * 2;
  size_t total = (size_t)CT * Ho * Wo;
  size_t i = (size_t)blockIdx.x * blockDim.x + threadIdx.x;
  if (i >= total) return;
  int wo = (int)(i % Wo);
  size_t t = i / Wo;
  int ho = (int)(t % Ho);
  int c  = (int)(t / Ho);
  float fx = (float)wo * ((float)(W - 1) / (float)(Wo - 1));
  float fy = (float)ho * ((float)(H - 1) / (float)(Ho - 1));
  out[i] = bilin_px(in + (size_t)c * H * W, H, W, fx, fy);
}

// ---------------------------------------------------------------------------
// Stage point selection: uncertainty over 3N points, top-nI via LDS bitonic
// grid = B blocks, 1024 threads, dyn LDS = P*8 bytes
// ---------------------------------------------------------------------------
__global__ __launch_bounds__(1024) void select_points(
    const float* __restrict__ temp,   // [B,8,Ht,Wt]
    const float* __restrict__ over,   // [B,nOver,2]
    const float* __restrict__ cov,    // [B,nCov,2]
    float* __restrict__ pts,          // [B,nI+nCov,2]
    int Ht, int Wt, int nOver, int nI, int nCov, int P) {
  extern __shared__ char sm[];
  float* key = (float*)sm;
  int*   idx = (int*)(sm + (size_t)P * sizeof(float));
  const int b = blockIdx.x;
  const int tid = threadIdx.x;
  const int HtWt = Ht * Wt;
  const float* T = temp + (size_t)b * 8 * HtWt;

  for (int i = tid; i < P; i += 1024) {
    float u = -1e30f;
    if (i < nOver) {
      float px = over[((size_t)b * nOver + i) * 2 + 0];
      float py = over[((size_t)b * nOver + i) * 2 + 1];
      float fx = px * (float)(Wt - 1);
      float fy = py * (float)(Ht - 1);
      int x0 = (int)floorf(fx);
      int y0 = (int)floorf(fy);
      x0 = x0 < 0 ? 0 : (x0 > Wt - 1 ? Wt - 1 : x0);
      y0 = y0 < 0 ? 0 : (y0 > Ht - 1 ? Ht - 1 : y0);
      int x1 = (x0 + 1 > Wt - 1) ? Wt - 1 : x0 + 1;
      int y1 = (y0 + 1 > Ht - 1) ? Ht - 1 : y0 + 1;
      float wx = fx - (float)x0, wy = fy - (float)y0;
      int   xs[2] = {x0, x1}, ys[2] = {y0, y1};
      float wxs[2] = {1.f - wx, wx}, wys[2] = {1.f - wy, wy};
      float pr[8];
#pragma unroll
      for (int c2 = 0; c2 < 8; ++c2) pr[c2] = 0.f;
      for (int cy = 0; cy < 2; ++cy) {
        for (int cx = 0; cx < 2; ++cx) {
          float w = wys[cy] * wxs[cx];
          const float* base = T + (size_t)ys[cy] * Wt + xs[cx];
          float v[8];
          float mx = -1e30f;
#pragma unroll
          for (int c2 = 0; c2 < 8; ++c2) {
            v[c2] = base[(size_t)c2 * HtWt];
            mx = fmaxf(mx, v[c2]);
          }
          float s = 0.f;
#pragma unroll
          for (int c2 = 0; c2 < 8; ++c2) { v[c2] = expf(v[c2] - mx); s += v[c2]; }
          float inv = w / s;
#pragma unroll
          for (int c2 = 0; c2 < 8; ++c2) pr[c2] += v[c2] * inv;
        }
      }
      float m1 = -1e30f, m2 = -1e30f;
#pragma unroll
      for (int c2 = 0; c2 < 8; ++c2) {
        float p = pr[c2];
        if (p > m1) { m2 = m1; m1 = p; } else if (p > m2) { m2 = p; }
      }
      u = m2 - m1;   // top2[1]-top2[0]; larger (nearer 0) = more uncertain
    }
    key[i] = u;
    idx[i] = i;
  }
  __syncthreads();

  // bitonic sort, descending by key
  for (int size = 2; size <= P; size <<= 1) {
    for (int stride = size >> 1; stride > 0; stride >>= 1) {
      for (int t = tid; t < (P >> 1); t += 1024) {
        int i = ((t & ~(stride - 1)) << 1) | (t & (stride - 1));
        int j = i + stride;
        bool desc = ((i & size) == 0);
        float ki = key[i], kj = key[j];
        if ((ki < kj) == desc) {
          key[i] = kj; key[j] = ki;
          int tt = idx[i]; idx[i] = idx[j]; idx[j] = tt;
        }
      }
      __syncthreads();
    }
  }

  const int N = nI + nCov;
  for (int t = tid; t < nI; t += 1024) {
    int s = idx[t];
    pts[((size_t)b * N + t) * 2 + 0] = over[((size_t)b * nOver + s) * 2 + 0];
    pts[((size_t)b * N + t) * 2 + 1] = over[((size_t)b * nOver + s) * 2 + 1];
  }
  for (int t = tid; t < nCov; t += 1024) {
    pts[((size_t)b * N + nI + t) * 2 + 0] = cov[((size_t)b * nCov + t) * 2 + 0];
    pts[((size_t)b * N + nI + t) * 2 + 1] = cov[((size_t)b * nCov + t) * 2 + 1];
  }
}

// ---------------------------------------------------------------------------
// Build X [M, Cpad] bf16 = concat(point_sample(temp), point_sample(fine)), pad 0
// grid = (M, ceil(Cpad/128)), block 128
// ---------------------------------------------------------------------------
__global__ __launch_bounds__(128) void build_X(
    const float* __restrict__ temp, int Ht, int Wt,
    const float* __restrict__ fine, int Cf, int Hf, int Wf,
    const float* __restrict__ pts, int N,
    unsigned short* __restrict__ X, int Cpad) {
  int m = blockIdx.x;
  int c = blockIdx.y * 128 + threadIdx.x;
  if (c >= Cpad) return;
  float px = pts[(size_t)m * 2 + 0];
  float py = pts[(size_t)m * 2 + 1];
  int b = m / N;
  float v = 0.f;
  if (c < 8) {
    v = bilin01(temp + ((size_t)b * 8 + c) * Ht * Wt, Ht, Wt, px, py);
  } else if (c < 8 + Cf) {
    v = bilin01(fine + ((size_t)b * Cf + (c - 8)) * Hf * Wf, Hf, Wf, px, py);
  }
  X[(size_t)m * Cpad + c] = f2bf(v);
}

// ---------------------------------------------------------------------------
// fp32 W [Nout, Cin] -> bf16 Wt [Npad, Kpad] (zero-padded)
// ---------------------------------------------------------------------------
__global__ __launch_bounds__(256) void convert_w(const float* __restrict__ W,
                                                 unsigned short* __restrict__ Wt,
                                                 int Nout, int Cin, int Npad, int Kpad) {
  size_t i = (size_t)blockIdx.x * 256 + threadIdx.x;
  size_t total = (size_t)Npad * Kpad;
  if (i >= total) return;
  int n = (int)(i / Kpad);
  int k = (int)(i % Kpad);
  float v = (n < Nout && k < Cin) ? W[(size_t)n * Cin + k] : 0.f;
  Wt[i] = f2bf(v);
}

// ---------------------------------------------------------------------------
// GEMM: Y = act(X[M,Kpad] @ Wt^T + bias) using v_wmma_f32_16x16x32_bf16.
// Block: 256 thr = 8 waves; each wave owns a 2Mx(NREP)N register tile
// (2*NREP accumulators, 2*NREP WMMAs per 32-K step; A fragments reused
// across the NREP B tiles -> A traffic / NREP).
// Block covers 256 rows x 16*NREP cols. B panel (16*NREP x Kpad) in LDS.
// grid = (M/256, Npad/(16*NREP)); dyn LDS = 16*NREP*Kpad*2 bytes.
// Ybf != null  -> bf16 out [M, yStride], relu optional.
// Ybf == null  -> fp32 out to d_out with [B,8,7168] mapping (Ns, outOff).
// ---------------------------------------------------------------------------
template <int NREP>
__global__ __launch_bounds__(256) void gemm_wmma_bf16(
    const unsigned short* __restrict__ X,
    const unsigned short* __restrict__ Wt,
    const float* __restrict__ bias,
    unsigned short* __restrict__ Ybf, int yStride,
    float* __restrict__ Yout,
    int M, int Kpad, int Nout, int doRelu, int Ns, int outOff) {
  extern __shared__ unsigned short Bpanel[];
  const int tid  = threadIdx.x;
  const int wave = tid >> 5;     // wave32
  const int lane = tid & 31;
  const int nrows = 16 * NREP;
  const int ncol0 = blockIdx.y * nrows;

  { // cooperative, coalesced B-panel load: 16*NREP rows of Wt
    const uint4* s4 = (const uint4*)(Wt + (size_t)ncol0 * Kpad);
    uint4* d4 = (uint4*)Bpanel;
    int n4 = (nrows * Kpad) >> 3;
    for (int i = tid; i < n4; i += 256) d4[i] = s4[i];
  }
  __syncthreads();

  const int mtile0 = (blockIdx.x * 8 + wave) * 2;
  const int mrow0  = mtile0 * 16 + (lane & 15);
  const int khalf  = (lane < 16) ? 0 : 8;    // A lane layout (16-bit A 16x32)
  const int bn     = lane & 15;
  const int bkhalf = (lane < 16) ? 0 : 8;    // B lane layout (K x 16)

  const unsigned short* a0p = X + (size_t)mrow0 * Kpad + khalf;
  const unsigned short* a1p = a0p + (size_t)16 * Kpad;
  const unsigned short* bp[NREP];
#pragma unroll
  for (int j = 0; j < NREP; ++j)
    bp[j] = Bpanel + ((size_t)j * 16 + bn) * Kpad + bkhalf;

  v8f acc[2][NREP];
#pragma unroll
  for (int j = 0; j < NREP; ++j) {
    acc[0][j] = (v8f){0.f, 0.f, 0.f, 0.f, 0.f, 0.f, 0.f, 0.f};
    acc[1][j] = acc[0][j];
  }

  for (int k = 0; k < Kpad; k += 32) {
    FragBF a0, a1, b[NREP];
    a0.q[0] = *(const uint4*)(a0p + k);
    a0.q[1] = *(const uint4*)(a0p + k + 16);
    a1.q[0] = *(const uint4*)(a1p + k);
    a1.q[1] = *(const uint4*)(a1p + k + 16);
#pragma unroll
    for (int j = 0; j < NREP; ++j) {
      b[j].q[0] = *(const uint4*)(bp[j] + k);
      b[j].q[1] = *(const uint4*)(bp[j] + k + 16);
    }
    if (k + 256 < Kpad) {          // -> global_prefetch_b8
      __builtin_prefetch(a0p + k + 256, 0, 1);
      __builtin_prefetch(a1p + k + 256, 0, 1);
    }
#pragma unroll
    for (int j = 0; j < NREP; ++j) {
      acc[0][j] = __builtin_amdgcn_wmma_f32_16x16x32_bf16(false, a0.bf, false, b[j].bf,
                                                          (short)0, acc[0][j], false, false);
      acc[1][j] = __builtin_amdgcn_wmma_f32_16x16x32_bf16(false, a1.bf, false, b[j].bf,
                                                          (short)0, acc[1][j], false, false);
    }
  }

  const int r0 = mtile0 * 16 + ((lane < 16) ? 0 : 8);
#pragma unroll
  for (int j = 0; j < NREP; ++j) {
    const int cbase = ncol0 + j * 16 + bn;
    const float bv = (cbase < Nout) ? bias[cbase] : 0.f;
#pragma unroll
    for (int r = 0; r < 8; ++r) {
      float v0 = acc[0][j][r] + bv;
      float v1 = acc[1][j][r] + bv;
      if (doRelu) { v0 = fmaxf(v0, 0.f); v1 = fmaxf(v1, 0.f); }
      int row0 = r0 + r, row1 = row0 + 16;
      if (Ybf) {
        Ybf[(size_t)row0 * yStride + cbase] = f2bf(v0);
        Ybf[(size_t)row1 * yStride + cbase] = f2bf(v1);
      } else if (cbase < Nout) {
        int b0 = row0 / Ns, n0 = row0 - b0 * Ns;
        int b1 = row1 / Ns, n1 = row1 - b1 * Ns;
        Yout[((size_t)b0 * 8 + cbase) * 7168 + outOff + n0] = v0;
        Yout[((size_t)b1 * 8 + cbase) * 7168 + outOff + n1] = v1;
      }
    }
  }
}

// ---------------------------------------------------------------------------
// Host orchestration
// ---------------------------------------------------------------------------
extern "C" void kernel_launch(void* const* d_in, const int* in_sizes, int n_in,
                              void* d_out, int out_size, void* d_ws, size_t ws_size,
                              hipStream_t stream) {
  (void)in_sizes; (void)n_in; (void)out_size; (void)ws_size;
  const float* refine = (const float*)d_in[0];
  const float* x0     = (const float*)d_in[1];
  const float* x1     = (const float*)d_in[2];
  const float* x2     = (const float*)d_in[3];
  const float* x3     = (const float*)d_in[4];
  const float* coarse = (const float*)d_in[5];
  // params flattened: mlp0(6..13), mlp1(14..21), mlp2(22..29), mlp3(30..37), mlpr(38..45)
  // each: W0,W1,W2,W3,b0,b1,b2,b3
  // rands: cov1..5 = 46..50, over1..5 = 51..55

  char* ws = (char*)d_ws;
  size_t off = 0;
  auto alloc = [&](size_t bytes) -> void* {
    void* p = ws + off;
    off += (bytes + 255) & ~(size_t)255;
    return p;
  };
  float* temp3 = (float*)alloc((size_t)4 * 8 * 128 * 128 * 4);
  float* temp4 = (float*)alloc((size_t)4 * 8 * 256 * 256 * 4);
  float* temp5 = (float*)alloc((size_t)4 * 8 * 512 * 512 * 4);
  float* pts   = (float*)alloc((size_t)4 * 2048 * 2 * 4);
  unsigned short* Xb  = (unsigned short*)alloc((size_t)8192 * 1056 * 2); // max M*Cpad
  unsigned short* Y1  = (unsigned short*)alloc((size_t)8192 * 512 * 2);
  unsigned short* Y2  = (unsigned short*)alloc((size_t)8192 * 512 * 2);
  unsigned short* Y3  = (unsigned short*)alloc((size_t)8192 * 512 * 2);
  unsigned short* Wt0 = (unsigned short*)alloc((size_t)512 * 1056 * 2);
  unsigned short* Wt1 = (unsigned short*)alloc((size_t)512 * 512 * 2);
  unsigned short* Wt2 = (unsigned short*)alloc((size_t)512 * 512 * 2);
  unsigned short* Wt3 = (unsigned short*)alloc((size_t)16 * 512 * 2);

  // chained align-corners upsampling of coarse logits
  {
    size_t t3 = (size_t)32 * 128 * 128;
    size_t t4 = (size_t)32 * 256 * 256;
    size_t t5 = (size_t)32 * 512 * 512;
    upsample2x<<<dim3((unsigned)((t3 + 255) / 256)), 256, 0, stream>>>(coarse, temp3, 32, 64, 64);
    upsample2x<<<dim3((unsigned)((t4 + 255) / 256)), 256, 0, stream>>>(temp3, temp4, 32, 128, 128);
    upsample2x<<<dim3((unsigned)((t5 + 255) / 256)), 256, 0, stream>>>(temp4, temp5, 32, 256, 256);
  }

  struct SC {
    const float* temp; int Ht, Wt;
    const float* fine; int Cf, Hf, Wf;
    int overIdx, covIdx;
    int nOver, nI, nCov, N, P;
    int mlpBase, outOff, Cpad;
  };
  const SC stages[5] = {
    { coarse,  64,  64, x3,     1024,  32,  32, 51, 46, 1536,  384, 128,  512, 2048, 30,    0, 1056 },
    { coarse,  64,  64, x2,      512,  64,  64, 52, 47, 1536,  384, 128,  512, 2048, 22,  512,  544 },
    { temp3,  128, 128, x1,      256, 128, 128, 53, 48, 6144, 1536, 512, 2048, 8192, 14, 1024,  288 },
    { temp4,  256, 256, x0,       64, 256, 256, 54, 49, 6144, 1536, 512, 2048, 8192,  6, 3072,   96 },
    { temp5,  512, 512, refine,   64, 512, 512, 55, 50, 6144, 1536, 512, 2048, 8192, 38, 5120,   96 },
  };

  for (int s = 0; s < 5; ++s) {
    const SC& c = stages[s];
    const float* over = (const float*)d_in[c.overIdx];
    const float* cov  = (const float*)d_in[c.covIdx];
    const float* W0 = (const float*)d_in[c.mlpBase + 0];
    const float* W1 = (const float*)d_in[c.mlpBase + 1];
    const float* W2 = (const float*)d_in[c.mlpBase + 2];
    const float* W3 = (const float*)d_in[c.mlpBase + 3];
    const float* b0 = (const float*)d_in[c.mlpBase + 4];
    const float* b1 = (const float*)d_in[c.mlpBase + 5];
    const float* b2 = (const float*)d_in[c.mlpBase + 6];
    const float* b3 = (const float*)d_in[c.mlpBase + 7];
    const int Cin = 8 + c.Cf;
    const int M = 4 * c.N;

    // 1) uncertainty top-k point selection (1 block per batch image)
    select_points<<<dim3(4), 1024, (size_t)c.P * 8, stream>>>(
        c.temp, over, cov, pts, c.Ht, c.Wt, c.nOver, c.nI, c.nCov, c.P);

    // 2) gather point features -> bf16 X
    build_X<<<dim3(M, (c.Cpad + 127) / 128), 128, 0, stream>>>(
        c.temp, c.Ht, c.Wt, c.fine, c.Cf, c.Hf, c.Wf, pts, c.N, Xb, c.Cpad);

    // 3) weight conversion (fp32 -> padded bf16)
    {
      size_t t0 = (size_t)512 * c.Cpad;
      size_t t1 = (size_t)512 * 512;
      size_t t3 = (size_t)16 * 512;
      convert_w<<<dim3((unsigned)((t0 + 255) / 256)), 256, 0, stream>>>(W0, Wt0, 512, Cin, 512, c.Cpad);
      convert_w<<<dim3((unsigned)((t1 + 255) / 256)), 256, 0, stream>>>(W1, Wt1, 512, 512, 512, 512);
      convert_w<<<dim3((unsigned)((t1 + 255) / 256)), 256, 0, stream>>>(W2, Wt2, 512, 512, 512, 512);
      convert_w<<<dim3((unsigned)((t3 + 255) / 256)), 256, 0, stream>>>(W3, Wt3, 8, 512, 16, 512);
    }

    // 4) MLP via WMMA GEMMs
    const unsigned mb = (unsigned)(M / 256);
    // Layer 0: 2-wide N tiling if the 32-row B panel fits in 64KB dynamic LDS
    if (c.Cpad <= 1024) {
      gemm_wmma_bf16<2><<<dim3(mb, 16), 256, (size_t)32 * c.Cpad * 2, stream>>>(
          Xb, Wt0, b0, Y1, 512, nullptr, M, c.Cpad, 512, 1, 0, 0);
    } else {
      gemm_wmma_bf16<1><<<dim3(mb, 32), 256, (size_t)16 * c.Cpad * 2, stream>>>(
          Xb, Wt0, b0, Y1, 512, nullptr, M, c.Cpad, 512, 1, 0, 0);
    }
    // Hidden layers: 2Mx2N register tiles, 32KB LDS B panel
    gemm_wmma_bf16<2><<<dim3(mb, 16), 256, (size_t)32 * 512 * 2, stream>>>(
        Y1, Wt1, b1, Y2, 512, nullptr, M, 512, 512, 1, 0, 0);
    gemm_wmma_bf16<2><<<dim3(mb, 16), 256, (size_t)32 * 512 * 2, stream>>>(
        Y2, Wt2, b2, Y3, 512, nullptr, M, 512, 512, 1, 0, 0);
    // Final layer: Nout=8 (padded to one 16-col tile), fp32 out to d_out
    gemm_wmma_bf16<1><<<dim3(mb, 1), 256, (size_t)16 * 512 * 2, stream>>>(
        Y3, Wt3, b3, nullptr, 0, (float*)d_out, M, 512, 8, 0, c.N, c.outOff);
  }
}
